// DeformableTransformerEncoderLayer_73126113181726
// MI455X (gfx1250) — matrix operand order repaired and checked by reading, not compiled, not run on hardware
//
#include <hip/hip_runtime.h>

#define M_TOTAL 43520   // B * Lq
#define LQ      21760
#define DM      256
#define DFFN    1024

typedef __attribute__((ext_vector_type(8)))  unsigned short v8us;
typedef __attribute__((ext_vector_type(16))) unsigned short v16us;
typedef __attribute__((ext_vector_type(16))) __bf16        v16bf;
typedef __attribute__((ext_vector_type(8)))  float         v8f;

__device__ __forceinline__ unsigned short f32_to_bf16(float f) {
  union { float f; unsigned u; } v; v.f = f;
  unsigned r = v.u + 0x7FFFu + ((v.u >> 16) & 1u);   // round-to-nearest-even
  return (unsigned short)(r >> 16);
}
__device__ __forceinline__ int iclamp(int x, int lo, int hi) {
  return x < lo ? lo : (x > hi ? hi : x);
}

// A 16x32 bf16 fragment: lane group 0 K=0..15, group 1
// K=16..31; caller pre-biases pointer by half*16. Two 16B loads.
__device__ __forceinline__ v16bf load_afrag(const unsigned short* p) {
  v8us lo = *(const v8us*)p;
  v8us hi = *(const v8us*)(p + 8);
  return __builtin_bit_cast(v16bf,
      __builtin_shufflevector(lo, hi,
          0, 1, 2, 3, 4, 5, 6, 7, 8, 9, 10, 11, 12, 13, 14, 15));
}
// B 32x16 bf16 fragment: lanes 0-15 K=0..15, lanes 16-31 K=16..31; caller
// pre-biases pointer by half*16. One contiguous 32B load.
__device__ __forceinline__ v16bf load_bfrag(const unsigned short* p) {
  return __builtin_bit_cast(v16bf, *(const v16us*)p);
}

// ---------------------------------------------------------------------------
// Weight transpose + f32 -> bf16:  W[K][N] -> Wt[N][K]
// ---------------------------------------------------------------------------
__global__ void wt_convert_kernel(const float* __restrict__ W,
                                  unsigned short* __restrict__ Wt,
                                  int K, int N) {
  const int i = blockIdx.x * 256 + threadIdx.x;   // exact grid
  const int k = i / N;
  const int n = i - k * N;
  Wt[(size_t)n * K + k] = f32_to_bf16(W[i]);
}

// ---------------------------------------------------------------------------
// query = src + pos ; bf16 casts of src and query
// ---------------------------------------------------------------------------
__global__ void prep_kernel(const float* __restrict__ src,
                            const float* __restrict__ pos,
                            unsigned short* __restrict__ s16,
                            unsigned short* __restrict__ q16) {
  const int i = blockIdx.x * 256 + threadIdx.x;   // exact grid (M*256 threads)
  const float s = src[i];
  s16[i] = f32_to_bf16(s);
  q16[i] = f32_to_bf16(s + pos[i]);
}

// ---------------------------------------------------------------------------
// WMMA bf16 GEMM:  C[M][N] = A[M][K] * Bt[N][K]^T + bias, optional ReLU.
// K is a template constant so every fragment load uses immediate offsets
// (single s_clause per K-step, no address chains).
// Wave computes a 32x64 tile: 8 x v_wmma_f32_16x16x32_bf16 per K-step,
// software-pipelined against the next step's 6 fragment loads.
// grid = (M/256, N/64), block = 256 (8 waves).
// ---------------------------------------------------------------------------
template <int K>
__global__ __launch_bounds__(256)
void gemm_bf16_wmma(const unsigned short* __restrict__ A,
                    const unsigned short* __restrict__ Bt,
                    const float* __restrict__ bias,
                    float* __restrict__ Cf,
                    unsigned short* __restrict__ Cb,
                    int M, int N, int relu) {
  const int lane = threadIdx.x & 31;
  const int wave = threadIdx.x >> 5;
  const int m0 = (blockIdx.x * 8 + wave) * 32;
  const int n0 = blockIdx.y * 64;
  if (m0 >= M) return;

  const int half = lane >> 4;   // 0: lanes 0-15, 1: lanes 16-31
  const int l15  = lane & 15;

  v8f acc[2][4] = {};

  const unsigned short* arow0 = A + (size_t)(m0 + l15) * K + half * 16;
  const unsigned short* arow1 = arow0 + (size_t)16 * K;
  const unsigned short* bcol  = Bt + (size_t)(n0 + l15) * K + half * 16;

  // prologue: fragments for k-step 0
  v16bf a0 = load_afrag(arow0);
  v16bf a1 = load_afrag(arow1);
  v16bf b0 = load_bfrag(bcol + 0 * 16 * K);
  v16bf b1 = load_bfrag(bcol + 1 * 16 * K);
  v16bf b2 = load_bfrag(bcol + 2 * 16 * K);
  v16bf b3 = load_bfrag(bcol + 3 * 16 * K);

  constexpr int NK = K / 32;
#pragma unroll
  for (int t = 0; t < NK; ++t) {
    const int kn = (t + 1) * 32;
    v16bf a0n = a0, a1n = a1, b0n = b0, b1n = b1, b2n = b2, b3n = b3;
    if (kn < K) {  // static per unrolled iteration
      a0n = load_afrag(arow0 + kn);
      a1n = load_afrag(arow1 + kn);
      b0n = load_bfrag(bcol + 0 * 16 * K + kn);
      b1n = load_bfrag(bcol + 1 * 16 * K + kn);
      b2n = load_bfrag(bcol + 2 * 16 * K + kn);
      b3n = load_bfrag(bcol + 3 * 16 * K + kn);
      if (kn + 32 < K) __builtin_prefetch(arow0 + kn + 32, 0, 3);
    }
    acc[0][0] = __builtin_amdgcn_wmma_f32_16x16x32_bf16(
        false, a0, false, b0, (short)0, acc[0][0], false, false);
    acc[0][1] = __builtin_amdgcn_wmma_f32_16x16x32_bf16(
        false, a0, false, b1, (short)0, acc[0][1], false, false);
    acc[0][2] = __builtin_amdgcn_wmma_f32_16x16x32_bf16(
        false, a0, false, b2, (short)0, acc[0][2], false, false);
    acc[0][3] = __builtin_amdgcn_wmma_f32_16x16x32_bf16(
        false, a0, false, b3, (short)0, acc[0][3], false, false);
    acc[1][0] = __builtin_amdgcn_wmma_f32_16x16x32_bf16(
        false, a1, false, b0, (short)0, acc[1][0], false, false);
    acc[1][1] = __builtin_amdgcn_wmma_f32_16x16x32_bf16(
        false, a1, false, b1, (short)0, acc[1][1], false, false);
    acc[1][2] = __builtin_amdgcn_wmma_f32_16x16x32_bf16(
        false, a1, false, b2, (short)0, acc[1][2], false, false);
    acc[1][3] = __builtin_amdgcn_wmma_f32_16x16x32_bf16(
        false, a1, false, b3, (short)0, acc[1][3], false, false);
    a0 = a0n; a1 = a1n; b0 = b0n; b1 = b1n; b2 = b2n; b3 = b3n;
  }

  // C/D layout: lane -> col n0+j*16+l15 ; VGPR r -> row base + 8*half + r
#pragma unroll
  for (int j = 0; j < 4; ++j) {
    const int col = n0 + j * 16 + l15;
    const float bv = bias ? bias[col] : 0.f;
#pragma unroll
    for (int h2 = 0; h2 < 2; ++h2) {
#pragma unroll
      for (int r = 0; r < 8; ++r) {
        const int row = m0 + h2 * 16 + half * 8 + r;
        float v = acc[h2][j][r] + bv;
        if (relu) v = fmaxf(v, 0.f);
        if (Cf) Cf[(size_t)row * N + col] = v;
        if (Cb) Cb[(size_t)row * N + col] = f32_to_bf16(v);
      }
    }
  }
}

// ---------------------------------------------------------------------------
// Fused softmax + deformable bilinear sampling.
// One wave per (query-row, head); lane = channel (HEAD_DIM = 32).
// grid = M_TOTAL blocks of 256 (8 heads per block).
// ---------------------------------------------------------------------------
__global__ __launch_bounds__(256)
void deform_attn_kernel(const float* __restrict__ value,     // [M, 8, 32]
                        const float* __restrict__ off,       // [M, 8,4,4,2]
                        const float* __restrict__ logits,    // [M, 8, 16]
                        const float* __restrict__ ref,       // [M, 4, 2]
                        unsigned short* __restrict__ out16)  // [M, 256] bf16
{
  const int lane = threadIdx.x & 31;
  const int h    = threadIdx.x >> 5;
  const int row  = blockIdx.x;          // b*LQ + q
  const int b    = row / LQ;

  // softmax over 16 (level,point) logits -- uniform across lanes
  const float* lg = logits + (size_t)row * 128 + h * 16;
  float w[16];
  float mx = -3.4e38f;
#pragma unroll
  for (int i = 0; i < 16; ++i) { w[i] = lg[i]; mx = fmaxf(mx, w[i]); }
  float s = 0.f;
#pragma unroll
  for (int i = 0; i < 16; ++i) { w[i] = __expf(w[i] - mx); s += w[i]; }
  const float inv = 1.f / s;

  const float* op = off + (size_t)row * 256 + h * 32;
  const float* rp = ref + (size_t)row * 8;

  const int HW[4] = {128, 64, 32, 16};              // square levels
  const int ST[4] = {0, 16384, 20480, 21504};

  float acc = 0.f;
#pragma unroll
  for (int l = 0; l < 4; ++l) {
    const int   Wl = HW[l];
    const float fS = (float)Wl;
    const float rx = rp[l * 2 + 0] * fS;
    const float ry = rp[l * 2 + 1] * fS;
    const int base = b * LQ + ST[l];
#pragma unroll
    for (int p = 0; p < 4; ++p) {
      // loc*W - 0.5 = ref*W + off - 0.5  (norm = (W,H) cancels)
      const float x = rx + op[(l * 4 + p) * 2 + 0] - 0.5f;
      const float y = ry + op[(l * 4 + p) * 2 + 1] - 0.5f;
      const float x0f = floorf(x), y0f = floorf(y);
      const float lx = x - x0f, ly = y - y0f;
      const int x0 = (int)x0f, y0 = (int)y0f;
      const float aw = w[l * 4 + p] * inv;
#pragma unroll
      for (int c = 0; c < 4; ++c) {
        const int dx = c & 1, dy = c >> 1;
        const int xc = x0 + dx, yc = y0 + dy;
        const float wt = (dx ? lx : 1.f - lx) * (dy ? ly : 1.f - ly) * aw;
        const bool ok = (xc >= 0) && (xc < Wl) && (yc >= 0) && (yc < Wl);
        const int xi = iclamp(xc, 0, Wl - 1);
        const int yi = iclamp(yc, 0, Wl - 1);
        const float v =
            value[((size_t)(base + yi * Wl + xi) * 8 + h) * 32 + lane];
        acc += ok ? wt * v : 0.f;
      }
    }
  }
  out16[(size_t)row * 256 + h * 32 + lane] = f32_to_bf16(acc);
}

// ---------------------------------------------------------------------------
// Fused residual + LayerNorm over 256 channels. One wave per row.
// Emits f32 and/or bf16. grid = M/8, block = 256.
// ---------------------------------------------------------------------------
__global__ __launch_bounds__(256)
void ln_fused_kernel(const float* __restrict__ a,
                     const float* __restrict__ bsrc,
                     const float* __restrict__ g,
                     const float* __restrict__ be,
                     float* __restrict__ of,
                     unsigned short* __restrict__ ob) {
  const int lane = threadIdx.x & 31;
  const int row  = blockIdx.x * 8 + (threadIdx.x >> 5);
  const float* ap = a    + (size_t)row * 256;
  const float* bp = bsrc + (size_t)row * 256;
  float x[8];
  float s = 0.f;
#pragma unroll
  for (int j = 0; j < 8; ++j) {
    const int i = lane + 32 * j;
    x[j] = ap[i] + bp[i];
    s += x[j];
  }
#pragma unroll
  for (int o2 = 16; o2 > 0; o2 >>= 1) s += __shfl_xor(s, o2, 32);
  const float mean = s * (1.f / 256.f);
  float v = 0.f;
#pragma unroll
  for (int j = 0; j < 8; ++j) { const float d = x[j] - mean; v += d * d; }
#pragma unroll
  for (int o2 = 16; o2 > 0; o2 >>= 1) v += __shfl_xor(v, o2, 32);
  const float rs = rsqrtf(v * (1.f / 256.f) + 1e-5f);
#pragma unroll
  for (int j = 0; j < 8; ++j) {
    const int i = lane + 32 * j;
    const float y = (x[j] - mean) * rs * g[i] + be[i];
    if (of) of[(size_t)row * 256 + i] = y;
    if (ob) ob[(size_t)row * 256 + i] = f32_to_bf16(y);
  }
}

// ---------------------------------------------------------------------------
extern "C" void kernel_launch(void* const* d_in, const int* in_sizes, int n_in,
                              void* d_out, int out_size, void* d_ws,
                              size_t ws_size, hipStream_t stream) {
  (void)in_sizes; (void)n_in; (void)out_size;
  const float* src    = (const float*)d_in[0];
  const float* pos    = (const float*)d_in[1];
  const float* ref    = (const float*)d_in[2];
  // d_in[3] spatial_shapes, d_in[4] level_start_index: static, hardcoded.
  const float* W_off  = (const float*)d_in[5];
  const float* b_off  = (const float*)d_in[6];
  const float* W_attn = (const float*)d_in[7];
  const float* b_attn = (const float*)d_in[8];
  const float* W_val  = (const float*)d_in[9];
  const float* b_val  = (const float*)d_in[10];
  const float* W_out  = (const float*)d_in[11];
  const float* b_out  = (const float*)d_in[12];
  const float* ln1_g  = (const float*)d_in[13];
  const float* ln1_b  = (const float*)d_in[14];
  const float* W1     = (const float*)d_in[15];
  const float* b1     = (const float*)d_in[16];
  const float* W2     = (const float*)d_in[17];
  const float* b2     = (const float*)d_in[18];
  const float* ln2_g  = (const float*)d_in[19];
  const float* ln2_b  = (const float*)d_in[20];
  float* out = (float*)d_out;

  const size_t M = M_TOTAL;
  char* ws = (char*)d_ws;
  size_t o = 0;
  auto alloc = [&](size_t bytes) -> void* {
    void* p = ws + o;
    o += (bytes + 255) & ~(size_t)255;
    return p;
  };

  unsigned short* s16   = (unsigned short*)alloc(M * 256 * 2);
  unsigned short* q16   = (unsigned short*)alloc(M * 256 * 2);   // adjacent to s16
  float*          val   = (float*)alloc(M * 256 * 4);
  float*          offb  = (float*)alloc(M * 256 * 4);
  float*          logi  = (float*)alloc(M * 128 * 4);
  unsigned short* ao16  = (unsigned short*)alloc(M * 256 * 2);
  unsigned short* h16   = (unsigned short*)alloc(M * 1024 * 2);
  unsigned short* wt_val  = (unsigned short*)alloc(256 * 256 * 2);
  unsigned short* wt_off  = (unsigned short*)alloc(256 * 256 * 2);
  unsigned short* wt_attn = (unsigned short*)alloc(128 * 256 * 2);
  unsigned short* wt_out  = (unsigned short*)alloc(256 * 256 * 2);
  unsigned short* wt_1    = (unsigned short*)alloc(1024 * 256 * 2);
  unsigned short* wt_2    = (unsigned short*)alloc(256 * 1024 * 2);
  if (o > ws_size) return;  // workspace too small; bail out safely

  // Dead-buffer reuse (exact-size overlays, lifetimes disjoint):
  float*          src2 = (float*)s16;            // spans s16+q16 (M*256*4)
  float*          x1   = (float*)val;            // value dead after sampling
  unsigned short* x116 = (unsigned short*)logi;  // logits dead after sampling
  float*          ffn  = (float*)offb;           // offsets dead after sampling

  const dim3 blk(256);
  const int MB = (int)(M / 256);  // 170 M-tiles of 256 rows (32 per wave)

  // 1) weights -> transposed bf16
  wt_convert_kernel<<<256 * 256 / 256, blk, 0, stream>>>(W_val, wt_val, 256, 256);
  wt_convert_kernel<<<256 * 256 / 256, blk, 0, stream>>>(W_off, wt_off, 256, 256);
  wt_convert_kernel<<<256 * 128 / 256, blk, 0, stream>>>(W_attn, wt_attn, 256, 128);
  wt_convert_kernel<<<256 * 256 / 256, blk, 0, stream>>>(W_out, wt_out, 256, 256);
  wt_convert_kernel<<<256 * 1024 / 256, blk, 0, stream>>>(W1, wt_1, 256, 1024);
  wt_convert_kernel<<<1024 * 256 / 256, blk, 0, stream>>>(W2, wt_2, 1024, 256);

  // 2) query = src + pos; bf16 casts
  prep_kernel<<<(unsigned)M, blk, 0, stream>>>(src, pos, s16, q16);

  // 3) projections (WMMA GEMMs)
  gemm_bf16_wmma<256><<<dim3(MB, 4), blk, 0, stream>>>(s16, wt_val, b_val, val,
                                                       nullptr, (int)M, 256, 0);
  gemm_bf16_wmma<256><<<dim3(MB, 4), blk, 0, stream>>>(q16, wt_off, b_off, offb,
                                                       nullptr, (int)M, 256, 0);
  gemm_bf16_wmma<256><<<dim3(MB, 2), blk, 0, stream>>>(q16, wt_attn, b_attn, logi,
                                                       nullptr, (int)M, 128, 0);

  // 4) fused softmax + bilinear deformable sampling
  deform_attn_kernel<<<(unsigned)M, blk, 0, stream>>>(val, offb, logi, ref, ao16);

  // 5) output projection, residual + LN1
  gemm_bf16_wmma<256><<<dim3(MB, 4), blk, 0, stream>>>(ao16, wt_out, b_out, src2,
                                                       nullptr, (int)M, 256, 0);
  ln_fused_kernel<<<(unsigned)(M / 8), blk, 0, stream>>>(src, src2, ln1_g, ln1_b,
                                                         x1, x116);

  // 6) FFN (ReLU fused in first GEMM epilogue), residual + LN2 -> d_out
  gemm_bf16_wmma<256><<<dim3(MB, 16), blk, 0, stream>>>(x116, wt_1, b1, nullptr,
                                                        h16, (int)M, 1024, 1);
  gemm_bf16_wmma<1024><<<dim3(MB, 4), blk, 0, stream>>>(h16, wt_2, b2, ffn,
                                                        nullptr, (int)M, 256, 0);
  ln_fused_kernel<<<(unsigned)(M / 8), blk, 0, stream>>>(x1, ffn, ln2_g, ln2_b,
                                                         out, nullptr);
}
